// GCN2_49289044689246
// MI455X (gfx1250) — matrix operand order repaired and checked by reading, not compiled, not run on hardware
//
#include <hip/hip_runtime.h>
#include <math.h>

#define NN     50000
#define EE     800000
#define INC    500
#define HID    64
#define OUTC   70
#define LAYERS 16

typedef __attribute__((ext_vector_type(16))) __bf16 v16bf;
typedef __attribute__((ext_vector_type(8)))  float  v8f;

static __device__ __forceinline__ __bf16 f2bf(float f) {
    unsigned int u = __builtin_bit_cast(unsigned int, f);
    u += 0x7FFFu + ((u >> 16) & 1u);                 // round-to-nearest-even
    return __builtin_bit_cast(__bf16, (unsigned short)(u >> 16));
}

// pack 8 consecutive-K floats (two float4s) into A-fragment slots [base..base+7]
static __device__ __forceinline__ void pack8(v16bf& a, int base, float4 p0, float4 p1) {
    a[base + 0] = f2bf(p0.x); a[base + 1] = f2bf(p0.y);
    a[base + 2] = f2bf(p0.z); a[base + 3] = f2bf(p0.w);
    a[base + 4] = f2bf(p1.x); a[base + 5] = f2bf(p1.y);
    a[base + 6] = f2bf(p1.z); a[base + 7] = f2bf(p1.w);
}

static __device__ __forceinline__ float4 blend4(float4 a, float4 b) {
    float4 r;
    r.x = 0.9f * a.x + 0.1f * b.x;  r.y = 0.9f * a.y + 0.1f * b.y;
    r.z = 0.9f * a.z + 0.1f * b.z;  r.w = 0.9f * a.w + 0.1f * b.w;
    return r;
}

// ---------------------------------------------------------------------------
// Kernel 1: h = x0 = relu(x @ W0 + b0)   [50000,500] @ [500,64]
// W0 in LDS transposed + K-padded to 512 -> unguarded ds_load_b128 B-frags.
// A-frags: unguarded global float4 pairs for k<480, clamped+select remainder.
// Epilogue also zeroes agg for the first SpMM.
// ---------------------------------------------------------------------------
__global__ __launch_bounds__(256) void k_in_gemm(const float* __restrict__ x,
                                                 const float* __restrict__ W0,
                                                 const float* __restrict__ b0,
                                                 float* __restrict__ h,
                                                 float* __restrict__ x0,
                                                 float* __restrict__ agg)
{
    __shared__ __bf16 sWt[HID * 512] __attribute__((aligned(32)));   // [n][k]
    for (int i = threadIdx.x; i < HID * 512; i += 256) {
        int k = i >> 6, n = i & 63;                   // i = k*64+n -> W0 coalesced
        sWt[n * 512 + k] = f2bf((k < INC) ? W0[i] : 0.0f);
    }
    __syncthreads();

    const int lane    = threadIdx.x & 31;
    const int wave    = threadIdx.x >> 5;
    const int rowBase = blockIdx.x * 128 + wave * 16;

    int arow = rowBase + (lane & 15);
    if (arow > NN - 1) arow = NN - 1;                 // clamp; stores masked below
    const float*  xr  = x + (size_t)arow * INC;
    const float4* xr4 = (const float4*)xr;            // row offset 2000B: 16B aligned
    const int khalf   = (lane >> 4) << 4;             // 0 or 16
    const int bcol    = lane & 15;

    v8f acc[4] = {};

    for (int kb = 0; kb < 480; kb += 32) {
        const int kA = kb + khalf;                    // multiple of 8
        float4 p0 = xr4[(kA >> 2) + 0], p1 = xr4[(kA >> 2) + 1];
        float4 q0 = xr4[(kA >> 2) + 2], q1 = xr4[(kA >> 2) + 3];   // kA+8
        v16bf a;
        pack8(a, 0, p0, p1);
        pack8(a, 8, q0, q1);

        const int kbB = kb + ((lane >> 4) << 4);
        #pragma unroll
        for (int t = 0; t < 4; ++t) {
            v16bf b = *(const v16bf*)(sWt + (t * 16 + bcol) * 512 + kbB);
            acc[t] = __builtin_amdgcn_wmma_f32_16x16x32_bf16(
                false, a, false, b, (short)0, acc[t], false, false);
        }
    }
    {   // remainder chunk kb=480 (k in [480,512), valid k<500):
        // clamped unconditional loads + value select (cndmask, no exec branches)
        v16bf a;
        #pragma unroll
        for (int v = 0; v < 8; ++v) {
            int k0 = 480 + (v << 1) + khalf;
            int kc = (k0 < INC - 1) ? k0 : (INC - 2);
            float f0 = xr[kc], f1 = xr[kc + 1];
            a[2 * v]     = f2bf((k0     < INC) ? f0 : 0.0f);
            a[2 * v + 1] = f2bf((k0 + 1 < INC) ? f1 : 0.0f);
        }
        const int kbB = 480 + ((lane >> 4) << 4);
        #pragma unroll
        for (int t = 0; t < 4; ++t) {
            v16bf b = *(const v16bf*)(sWt + (t * 16 + bcol) * 512 + kbB);
            acc[t] = __builtin_amdgcn_wmma_f32_16x16x32_bf16(
                false, a, false, b, (short)0, acc[t], false, false);
        }
    }

    #pragma unroll
    for (int r = 0; r < 8; ++r) {
        int row = rowBase + r + ((lane >> 4) << 3);
        if (row < NN) {
            #pragma unroll
            for (int t = 0; t < 4; ++t) {
                int col = t * 16 + (lane & 15);
                size_t idx = (size_t)row * HID + col;
                float v = fmaxf(acc[t][r] + b0[col], 0.0f);
                h[idx]   = v;
                x0[idx]  = v;
                agg[idx] = 0.0f;                      // prime agg for layer 0 SpMM
            }
        }
    }
}

// ---------------------------------------------------------------------------
// Kernel 2: SpMM scatter: agg[row[e]] += w[e] * h[col[e]]
// 16 threads per edge, float4 gather + 4 fp32 atomics (L2-resident).
// ---------------------------------------------------------------------------
__global__ __launch_bounds__(256) void k_spmm(const long long* __restrict__ rows,
                                              const long long* __restrict__ cols,
                                              const float* __restrict__ w,
                                              const float* __restrict__ h,
                                              float* __restrict__ agg)
{
    long long gid = (long long)blockIdx.x * blockDim.x + threadIdx.x;
    long long e   = gid >> 4;
    if (e >= EE) return;
    int f = (int)(gid & 15) << 2;
    long long r = rows[e];
    long long c = cols[e];
    float we = w[e];
    const float4 hv = *(const float4*)(h + c * HID + f);
    float* ap = agg + r * HID + f;
    atomicAdd(ap + 0, we * hv.x);
    atomicAdd(ap + 1, we * hv.y);
    atomicAdd(ap + 2, we * hv.z);
    atomicAdd(ap + 3, we * hv.w);
}

// ---------------------------------------------------------------------------
// Kernel 3: layer update, fused:
//   t = 0.9*agg + 0.1*x0 ; h = relu((1-beta)*t + beta*(t @ W)) ; agg = 0
// W (64x64) in LDS transposed; A-frags via float4 loads + fused blend.
// ---------------------------------------------------------------------------
__global__ __launch_bounds__(256) void k_layer(float* __restrict__ agg,
                                               const float* __restrict__ x0,
                                               const float* __restrict__ W,
                                               float* __restrict__ h,
                                               float beta)
{
    __shared__ __bf16 sWt[HID * HID] __attribute__((aligned(32)));   // [n][k]
    for (int i = threadIdx.x; i < HID * HID; i += 256) {
        int k = i >> 6, n = i & 63;
        sWt[n * 64 + k] = f2bf(W[i]);
    }
    __syncthreads();

    const int lane    = threadIdx.x & 31;
    const int wave    = threadIdx.x >> 5;
    const int rowBase = blockIdx.x * 128 + wave * 16;

    int arow = rowBase + (lane & 15);
    if (arow > NN - 1) arow = NN - 1;
    const float4* ar4 = (const float4*)(agg + (size_t)arow * HID);
    const float4* pr4 = (const float4*)(x0  + (size_t)arow * HID);
    const int khalf   = (lane >> 4) << 4;
    const int bcol    = lane & 15;

    v8f acc[4] = {};

    #pragma unroll
    for (int kb = 0; kb < HID; kb += 32) {
        const int kA = kb + khalf;
        float4 t0 = blend4(ar4[(kA >> 2) + 0], pr4[(kA >> 2) + 0]);
        float4 t1 = blend4(ar4[(kA >> 2) + 1], pr4[(kA >> 2) + 1]);
        float4 t2 = blend4(ar4[(kA >> 2) + 2], pr4[(kA >> 2) + 2]);
        float4 t3 = blend4(ar4[(kA >> 2) + 3], pr4[(kA >> 2) + 3]);
        v16bf a;
        pack8(a, 0, t0, t1);
        pack8(a, 8, t2, t3);

        const int kbB = kb + ((lane >> 4) << 4);
        #pragma unroll
        for (int t = 0; t < 4; ++t) {
            v16bf b = *(const v16bf*)(sWt + (t * 16 + bcol) * 64 + kbB);
            acc[t] = __builtin_amdgcn_wmma_f32_16x16x32_bf16(
                false, a, false, b, (short)0, acc[t], false, false);
        }
    }

    #pragma unroll
    for (int r = 0; r < 8; ++r) {
        int row = rowBase + r + ((lane >> 4) << 3);
        if (row < NN) {
            #pragma unroll
            for (int t = 0; t < 4; ++t) {
                int col = t * 16 + (lane & 15);
                size_t idx = (size_t)row * HID + col;
                float tv = 0.9f * agg[idx] + 0.1f * x0[idx];
                float o  = (1.0f - beta) * tv + beta * acc[t][r];
                h[idx]   = fmaxf(o, 0.0f);
                agg[idx] = 0.0f;                      // ready for next layer's SpMM
            }
        }
    }
}

// ---------------------------------------------------------------------------
// Kernel 4: logits = h @ W1 + b1   [50000,64] @ [64,70], N padded to 80 (5 tiles)
// ---------------------------------------------------------------------------
__global__ __launch_bounds__(256) void k_out_gemm(const float* __restrict__ h,
                                                  const float* __restrict__ W1,
                                                  const float* __restrict__ b1,
                                                  float* __restrict__ out)
{
    __shared__ __bf16 sWt[80 * HID] __attribute__((aligned(32)));    // [n][k]
    for (int i = threadIdx.x; i < 80 * HID; i += 256) {
        int k = i / 80, n = i % 80;
        sWt[n * 64 + k] = f2bf((n < OUTC) ? W1[k * OUTC + n] : 0.0f);
    }
    __syncthreads();

    const int lane    = threadIdx.x & 31;
    const int wave    = threadIdx.x >> 5;
    const int rowBase = blockIdx.x * 128 + wave * 16;

    int arow = rowBase + (lane & 15);
    if (arow > NN - 1) arow = NN - 1;
    const float4* hr4 = (const float4*)(h + (size_t)arow * HID);
    const int khalf   = (lane >> 4) << 4;
    const int bcol    = lane & 15;

    v8f acc[5] = {};

    #pragma unroll
    for (int kb = 0; kb < HID; kb += 32) {
        const int kA = kb + khalf;
        float4 p0 = hr4[(kA >> 2) + 0], p1 = hr4[(kA >> 2) + 1];
        float4 q0 = hr4[(kA >> 2) + 2], q1 = hr4[(kA >> 2) + 3];
        v16bf a;
        pack8(a, 0, p0, p1);
        pack8(a, 8, q0, q1);

        const int kbB = kb + ((lane >> 4) << 4);
        #pragma unroll
        for (int t = 0; t < 5; ++t) {
            v16bf b = *(const v16bf*)(sWt + (t * 16 + bcol) * 64 + kbB);
            acc[t] = __builtin_amdgcn_wmma_f32_16x16x32_bf16(
                false, a, false, b, (short)0, acc[t], false, false);
        }
    }

    #pragma unroll
    for (int r = 0; r < 8; ++r) {
        int row = rowBase + r + ((lane >> 4) << 3);
        if (row < NN) {
            #pragma unroll
            for (int t = 0; t < 5; ++t) {
                int col = t * 16 + (lane & 15);
                if (col < OUTC)
                    out[(size_t)row * OUTC + col] = acc[t][r] + b1[col];
            }
        }
    }
}

// ---------------------------------------------------------------------------
// Kernel 5: in-place log_softmax over 70 classes; one wave32 per row.
// ---------------------------------------------------------------------------
__global__ __launch_bounds__(256) void k_lsm(float* __restrict__ out)
{
    const int lane = threadIdx.x & 31;
    const int wave = threadIdx.x >> 5;
    const int row  = blockIdx.x * 8 + wave;
    if (row >= NN) return;
    float* p = out + (size_t)row * OUTC;

    float m = -3.0e38f;
    for (int j = lane; j < OUTC; j += 32) m = fmaxf(m, p[j]);
    #pragma unroll
    for (int off = 16; off > 0; off >>= 1) m = fmaxf(m, __shfl_xor(m, off, 32));

    float s = 0.0f;
    for (int j = lane; j < OUTC; j += 32) s += __expf(p[j] - m);
    #pragma unroll
    for (int off = 16; off > 0; off >>= 1) s += __shfl_xor(s, off, 32);

    const float lse = m + __logf(s);
    for (int j = lane; j < OUTC; j += 32) p[j] = p[j] - lse;
}

// ---------------------------------------------------------------------------
extern "C" void kernel_launch(void* const* d_in, const int* in_sizes, int n_in,
                              void* d_out, int out_size, void* d_ws, size_t ws_size,
                              hipStream_t stream)
{
    (void)in_sizes; (void)n_in; (void)out_size; (void)ws_size;

    const float*     x  = (const float*)d_in[0];
    const long long* ei = (const long long*)d_in[1];   // int64 edge_index [2,E]
    const float*     ew = (const float*)d_in[2];
    const float*     W0 = (const float*)d_in[3];
    const float*     b0 = (const float*)d_in[4];
    const float*     Wl = (const float*)d_in[5];
    const float*     W1 = (const float*)d_in[6];
    const float*     b1 = (const float*)d_in[7];
    float* out = (float*)d_out;

    float* x0  = (float*)d_ws;                         // N*64
    float* h   = x0 + (size_t)NN * HID;                // N*64
    float* agg = h  + (size_t)NN * HID;                // N*64

    const int gemmBlocks = (NN + 127) / 128;

    k_in_gemm<<<gemmBlocks, 256, 0, stream>>>(x, W0, b0, h, x0, agg);

    const long long spmmThreads = (long long)EE * 16;
    const unsigned  spmmBlocks  = (unsigned)((spmmThreads + 255) / 256);

    for (int l = 0; l < LAYERS; ++l) {
        float beta = logf(0.5f / (float)(l + 1) + 1.0f);
        k_spmm <<<spmmBlocks, 256, 0, stream>>>(ei, ei + EE, ew, h, agg);
        k_layer<<<gemmBlocks, 256, 0, stream>>>(agg, x0, Wl + (size_t)l * HID * HID, h, beta);
    }

    k_out_gemm<<<gemmBlocks, 256, 0, stream>>>(h, W1, b1, out);
    k_lsm    <<<(NN + 7) / 8, 256, 0, stream>>>(out);
}
